// FastRNNModelLayer_32873679684255
// MI455X (gfx1250) — matrix operand order repaired and checked by reading, not compiled, not run on hardware
//
#include <hip/hip_runtime.h>
#include <hip/hip_bf16.h>

// Problem sizes (fixed by the reference)
#define T_   256
#define B_   8
#define H_   8
#define D_   64
#define L_   2
#define HID_ 512
#define FF_  2048
#define IN_  512
#define TB_  (T_ * B_)        // 2048 rows
#define SEG_ (5 * D_ + 2)     // 322 features per head from slow_w
#define NS_  (H_ * SEG_)      // 2576
#define TSTRIDE_ (B_ * H_ * SEG_)   // 20608 floats per timestep

typedef __attribute__((ext_vector_type(16))) _Float16 v16h;
typedef __attribute__((ext_vector_type(8)))  _Float16 v8h;
typedef __attribute__((ext_vector_type(8)))  float    v8f;

// ---------------------------------------------------------------------------
// Tiled GEMM on CDNA5 WMMA: C[M,N] = A[M,K] @ W[N,K]^T (+bias)(+resid)(relu)
// One wave computes a 64(M) x 16(N) strip: 4 accumulators reuse each B
// fragment across 4 WMMAs (raises FLOP/byte from 8 to ~13 at the load
// interface). 8 waves per block. K-step 32 (f16 inputs, f32 accumulate).
// Lane layouts per CDNA5 ISA:
//   A: lane row = lane&15; elems 0..7 -> K=(lane>>4)*8+e, 8..15 -> 16+(lane>>4)*8+e
//   B: lane col = lane&15; elems 0..7 -> K=(lane>>4)*8+e, 8..15 -> 16+(lane>>4)*8+e (same as A)
//   C: lane col = lane&15; VGPR e     -> row e + 8*(lane>>4)
// ---------------------------------------------------------------------------
template <bool BIAS, bool RELU, bool RESID, bool OUTF32, bool OUTF16>
__global__ __launch_bounds__(256) void gemm_wmma_f16(
    const _Float16* __restrict__ A, const _Float16* __restrict__ W,
    const float* __restrict__ bias, float* __restrict__ Cf,
    _Float16* __restrict__ Ch, int M, int N, int K)
{
    const int lane   = threadIdx.x & 31;
    const int wid    = threadIdx.x >> 5;
    const int tilesN = N >> 4;
    const int tile   = blockIdx.x * 8 + wid;
    if (tile >= (M >> 6) * tilesN) return;   // wave-uniform exit (EXEC all-1 for WMMA)

    const int tm = (tile / tilesN) << 6;     // 64-row strip
    const int tn = (tile % tilesN) << 4;
    const int r  = lane & 15;
    const int kh = lane >> 4;

    const _Float16* __restrict__ arow0 = A + (size_t)(tm + r) * K;
    const _Float16* __restrict__ arow1 = arow0 + (size_t)16 * K;
    const _Float16* __restrict__ arow2 = arow0 + (size_t)32 * K;
    const _Float16* __restrict__ arow3 = arow0 + (size_t)48 * K;
    const _Float16* __restrict__ wrow  = W + (size_t)(tn + r) * K;

    v8f acc0 = {}, acc1 = {}, acc2 = {}, acc3 = {};
    for (int k0 = 0; k0 < K; k0 += 32) {
        v8h bl = *(const v8h*)(wrow + k0 + kh * 8);
        v8h bh = *(const v8h*)(wrow + k0 + 16 + kh * 8);
        v16h bv = __builtin_shufflevector(bl, bh, 0,1,2,3,4,5,6,7,8,9,10,11,12,13,14,15);
        if (k0 + 32 < K)
            __builtin_prefetch(wrow + k0 + 64, 0, 1);   // global_prefetch_b8

        v8h a0l = *(const v8h*)(arow0 + k0 + kh * 8);
        v8h a0h = *(const v8h*)(arow0 + k0 + 16 + kh * 8);
        v8h a1l = *(const v8h*)(arow1 + k0 + kh * 8);
        v8h a1h = *(const v8h*)(arow1 + k0 + 16 + kh * 8);
        v8h a2l = *(const v8h*)(arow2 + k0 + kh * 8);
        v8h a2h = *(const v8h*)(arow2 + k0 + 16 + kh * 8);
        v8h a3l = *(const v8h*)(arow3 + k0 + kh * 8);
        v8h a3h = *(const v8h*)(arow3 + k0 + 16 + kh * 8);

        v16h av0 = __builtin_shufflevector(a0l, a0h, 0,1,2,3,4,5,6,7,8,9,10,11,12,13,14,15);
        v16h av1 = __builtin_shufflevector(a1l, a1h, 0,1,2,3,4,5,6,7,8,9,10,11,12,13,14,15);
        v16h av2 = __builtin_shufflevector(a2l, a2h, 0,1,2,3,4,5,6,7,8,9,10,11,12,13,14,15);
        v16h av3 = __builtin_shufflevector(a3l, a3h, 0,1,2,3,4,5,6,7,8,9,10,11,12,13,14,15);

        acc0 = __builtin_amdgcn_wmma_f32_16x16x32_f16(false, av0, false, bv, (short)0, acc0, false, false);
        acc1 = __builtin_amdgcn_wmma_f32_16x16x32_f16(false, av1, false, bv, (short)0, acc1, false, false);
        acc2 = __builtin_amdgcn_wmma_f32_16x16x32_f16(false, av2, false, bv, (short)0, acc2, false, false);
        acc3 = __builtin_amdgcn_wmma_f32_16x16x32_f16(false, av3, false, bv, (short)0, acc3, false, false);
    }

    const int n = tn + r;
    const float bvv = BIAS ? bias[n] : 0.0f;
    v8f accs[4] = { acc0, acc1, acc2, acc3 };
    #pragma unroll
    for (int mt = 0; mt < 4; ++mt) {
        #pragma unroll
        for (int e = 0; e < 8; ++e) {
            const int m = tm + mt * 16 + e + (kh << 3);
            float v = accs[mt][e] + bvv;
            if (RESID) v += Cf[(size_t)m * N + n];
            if (RELU)  v = fmaxf(v, 0.0f);
            if (OUTF32) Cf[(size_t)m * N + n] = v;
            if (OUTF16) Ch[(size_t)m * N + n] = (_Float16)v;
        }
    }
}

// ---------------------------------------------------------------------------
// Row LayerNorm (C=512), f32 in -> f16 out (feeds the next WMMA GEMM)
// ---------------------------------------------------------------------------
__global__ __launch_bounds__(256) void layernorm_f16(
    const float* __restrict__ x, const float* __restrict__ g,
    const float* __restrict__ b, _Float16* __restrict__ y, int C)
{
    const int row = blockIdx.x;
    const float* xr = x + (size_t)row * C;
    __shared__ float rs[8], rs2[8], st[2];

    float s = 0.f, s2 = 0.f;
    for (int j = threadIdx.x; j < C; j += 256) { float v = xr[j]; s += v; s2 += v * v; }
    #pragma unroll
    for (int o = 16; o; o >>= 1) { s += __shfl_xor(s, o, 32); s2 += __shfl_xor(s2, o, 32); }
    if ((threadIdx.x & 31) == 0) { rs[threadIdx.x >> 5] = s; rs2[threadIdx.x >> 5] = s2; }
    __syncthreads();
    if (threadIdx.x == 0) {
        float S = 0.f, S2 = 0.f;
        #pragma unroll
        for (int w = 0; w < 8; ++w) { S += rs[w]; S2 += rs2[w]; }
        const float mean = S / C;
        st[0] = mean;
        st[1] = rsqrtf(fmaxf(S2 / C - mean * mean, 0.f) + 1e-5f);
    }
    __syncthreads();
    const float mean = st[0], inv = st[1];
    for (int j = threadIdx.x; j < C; j += 256)
        y[(size_t)row * C + j] = (_Float16)((xr[j] - mean) * inv * g[j] + b[j]);
}

// ---------------------------------------------------------------------------
// Per-(t,b,h) activation post-processing: softmax on q,k,rk; sigmoid on betas.
// 64 threads (2 waves), in-place on the qkvb buffer.
// ---------------------------------------------------------------------------
__device__ __forceinline__ void softmax64(float* __restrict__ seg, int j,
                                          float* __restrict__ red)
{
    float v = seg[j];
    float m = v;
    #pragma unroll
    for (int o = 16; o; o >>= 1) m = fmaxf(m, __shfl_xor(m, o, 32));
    if ((j & 31) == 0) red[j >> 5] = m;
    __syncthreads();
    m = fmaxf(red[0], red[1]);
    const float e = __expf(v - m);
    float s = e;
    #pragma unroll
    for (int o = 16; o; o >>= 1) s += __shfl_xor(s, o, 32);
    if ((j & 31) == 0) red[2 + (j >> 5)] = s;
    __syncthreads();
    seg[j] = e / (red[2] + red[3]);
    __syncthreads();
}

__global__ __launch_bounds__(64) void qkvb_post(float* __restrict__ qkvb)
{
    float* p = qkvb + (size_t)blockIdx.x * SEG_;
    __shared__ float red[4];
    const int j = threadIdx.x;
    softmax64(p,       j, red);   // q
    softmax64(p + 64,  j, red);   // k
    softmax64(p + 192, j, red);   // rk
    if (j < 2) { const float x = p[320 + j]; p[320 + j] = 1.f / (1.f + __expf(-x)); }
}

// ---------------------------------------------------------------------------
// Async staging of one timestep's 322 dwords into an LDS double buffer using
// CDNA5 GLOBAL_LOAD_ASYNC_TO_LDS_B32 (ASYNCcnt-tracked, fire-and-forget DMA).
// The generic pointer's low 32 bits are the wave-relative LDS byte offset
// (ISA 10.2: flat->LDS mapping truncates to addr[31:0]), which is exactly
// what the instruction's VDST VGPR carries.
// ---------------------------------------------------------------------------
__device__ __forceinline__ void stage_async(const float* __restrict__ gbase,
                                            unsigned bh, float* __restrict__ dst,
                                            int i)
{
    const unsigned long long sb64 = (unsigned long long)gbase;
    #pragma unroll
    for (int rpt = 0; rpt < 5; ++rpt) {
        const int idx = i + rpt * 64;
        const unsigned voff = (bh * (unsigned)SEG_ + (unsigned)idx) * 4u;
        const unsigned lds  = (unsigned)(size_t)(dst + idx);
        asm volatile("global_load_async_to_lds_b32 %0, %1, %2"
                     :: "v"(lds), "v"(voff), "s"(sb64) : "memory");
    }
    if (i < 2) {
        const int idx = 320 + i;
        const unsigned voff = (bh * (unsigned)SEG_ + (unsigned)idx) * 4u;
        const unsigned lds  = (unsigned)(size_t)(dst + idx);
        asm volatile("global_load_async_to_lds_b32 %0, %1, %2"
                     :: "v"(lds), "v"(voff), "s"(sb64) : "memory");
    }
}

// ---------------------------------------------------------------------------
// Fused delta-rule + fast-weight RNN scan. One block per (b,h); 64 threads.
// Each thread owns one row (64 f32) of W and R in REGISTERS (fully unrolled).
// The next timestep's q/k/v/rk/rv/beta vectors are DMA'd into an LDS double
// buffer with async-to-LDS while the current step's dependent VALU chain
// runs; each step fences with s_wait_asynccnt 0 + the workgroup barrier.
// ---------------------------------------------------------------------------
__global__ __launch_bounds__(64) void fastrnn_scan(
    const float* __restrict__ qkvb, const float* __restrict__ W0,
    const float* __restrict__ R0,   const float* __restrict__ h0,
    float* __restrict__ hseq)
{
    const int bh = blockIdx.x;   // b*H + h
    const int i  = threadIdx.x;  // row index 0..63

    float Wr[64], Rr[64];
    {
        const float* wsrc = W0 + (size_t)bh * 4096 + (size_t)i * 64;
        const float* rsrc = R0 + (size_t)bh * 4096 + (size_t)i * 64;
        #pragma unroll
        for (int j = 0; j < 64; ++j) { Wr[j] = wsrc[j]; Rr[j] = rsrc[j]; }
    }
    float hcar = h0[(size_t)bh * 64 + i];

    __shared__ float stage[2][SEG_ + 2];   // double-buffered timestep vectors
    __shared__ float sm[64];
    __shared__ float red[4];

    // Prologue: kick off DMA for t = 0.
    stage_async(qkvb, (unsigned)bh, &stage[0][0], i);

    for (int t = 0; t < T_; ++t) {
        asm volatile("s_wait_asynccnt 0x0" ::: "memory");
        __syncthreads();                        // stage[t&1] now valid everywhere
        const float* cur = &stage[t & 1][0];

        if (t + 1 < T_)                         // overlap next-step DMA with compute
            stage_async(qkvb + (size_t)(t + 1) * TSTRIDE_,
                        (unsigned)bh, &stage[(t + 1) & 1][0], i);

        const float bt  = cur[320];
        const float rbt = cur[321];

        // delta rule: v_old = W k ; W += beta (v - v_old) k^T ; z = W_new q
        float vo = 0.f;
        #pragma unroll
        for (int j = 0; j < 64; ++j) vo += Wr[j] * cur[64 + j];
        const float upd = bt * (cur[128 + i] - vo);
        float zi = 0.f;
        #pragma unroll
        for (int j = 0; j < 64; ++j) { Wr[j] += upd * cur[64 + j]; zi += Wr[j] * cur[j]; }

        // RNN fast-weight update
        float vo2 = 0.f;
        #pragma unroll
        for (int j = 0; j < 64; ++j) vo2 += Rr[j] * cur[192 + j];
        const float upd2 = rbt * (cur[256 + i] - vo2);

        // softmax over previous hidden state (one value per thread)
        float mx = hcar;
        #pragma unroll
        for (int o = 16; o; o >>= 1) mx = fmaxf(mx, __shfl_xor(mx, o, 32));
        if ((i & 31) == 0) red[i >> 5] = mx;
        __syncthreads();
        mx = fmaxf(red[0], red[1]);
        const float e = __expf(hcar - mx);
        float sum = e;
        #pragma unroll
        for (int o = 16; o; o >>= 1) sum += __shfl_xor(sum, o, 32);
        if ((i & 31) == 0) red[2 + (i >> 5)] = sum;
        __syncthreads();
        sm[i] = e / (red[2] + red[3]);
        __syncthreads();

        float hn = zi;
        #pragma unroll
        for (int j = 0; j < 64; ++j) { Rr[j] += upd2 * cur[192 + j]; hn += Rr[j] * sm[j]; }

        hseq[(size_t)t * (B_ * H_ * D_) + (size_t)bh * 64 + i] = hn;
        hcar = hn;
        __syncthreads();   // all reads of stage[t&1]/sm done before next overwrite
    }
}

// ---------------------------------------------------------------------------
// Elementwise helpers
// ---------------------------------------------------------------------------
__global__ void cvt_f32_f16_k(const float* __restrict__ x, _Float16* __restrict__ y, int n)
{ const int i = blockIdx.x * 256 + threadIdx.x; if (i < n) y[i] = (_Float16)x[i]; }

__global__ void copy_f32_k(const float* __restrict__ x, float* __restrict__ y, int n)
{ const int i = blockIdx.x * 256 + threadIdx.x; if (i < n) y[i] = x[i]; }

// ---------------------------------------------------------------------------
extern "C" void kernel_launch(void* const* d_in, const int* in_sizes, int n_in,
                              void* d_out, int out_size, void* d_ws, size_t ws_size,
                              hipStream_t stream)
{
    (void)in_sizes; (void)n_in; (void)out_size; (void)ws_size;

    const float* x       = (const float*)d_in[0];
    const float* fw      = (const float*)d_in[1];
    const float* rfw     = (const float*)d_in[2];
    const float* rnn     = (const float*)d_in[3];
    const float* proj_w  = (const float*)d_in[4];
    const float* proj_b  = (const float*)d_in[5];
    const float* ln1_g   = (const float*)d_in[6];
    const float* ln1_b   = (const float*)d_in[7];
    const float* slow_w  = (const float*)d_in[8];
    const float* out_w   = (const float*)d_in[9];
    const float* ff_ln_g = (const float*)d_in[10];
    const float* ff_ln_b = (const float*)d_in[11];
    const float* ff_w1   = (const float*)d_in[12];
    const float* ff_b1   = (const float*)d_in[13];
    const float* ff_w2   = (const float*)d_in[14];
    const float* ff_b2   = (const float*)d_in[15];

    // Workspace layout (~55 MB)
    char* ws = (char*)d_ws;
    size_t o = 0;
    float*    out_f = (float*)(ws + o);     o += (size_t)TB_ * HID_ * 4;
    _Float16* a16   = (_Float16*)(ws + o);  o += (size_t)TB_ * HID_ * 2;
    _Float16* ff16  = (_Float16*)(ws + o);  o += (size_t)TB_ * FF_ * 2;
    float*    qkvb  = (float*)(ws + o);     o += (size_t)TB_ * NS_ * 4;
    float*    hseq  = (float*)(ws + o);     o += (size_t)TB_ * HID_ * 4;
    _Float16* w16   = (_Float16*)(ws + o);
    size_t wo = 0;
    _Float16* proj16 = w16 + wo;  wo += (size_t)HID_ * IN_;
    _Float16* slow16 = w16 + wo;  wo += (size_t)L_ * NS_ * HID_;
    _Float16* outw16 = w16 + wo;  wo += (size_t)L_ * HID_ * HID_;
    _Float16* ffw116 = w16 + wo;  wo += (size_t)L_ * FF_ * HID_;
    _Float16* ffw216 = w16 + wo;  wo += (size_t)L_ * HID_ * FF_;

    auto cvt = [&](const float* src, _Float16* dst, size_t n) {
        cvt_f32_f16_k<<<dim3((unsigned)((n + 255) / 256)), dim3(256), 0, stream>>>(
            src, dst, (int)n);
    };
    auto gemm_grid = [](int M, int N) {
        return dim3((unsigned)(((M >> 6) * (N >> 4) + 7) / 8));
    };

    // One-time f32 -> f16 weight conversion (halves GEMM read bandwidth)
    cvt(x,      a16,    (size_t)TB_ * IN_);
    cvt(proj_w, proj16, (size_t)HID_ * IN_);
    cvt(slow_w, slow16, (size_t)L_ * NS_ * HID_);
    cvt(out_w,  outw16, (size_t)L_ * HID_ * HID_);
    cvt(ff_w1,  ffw116, (size_t)L_ * FF_ * HID_);
    cvt(ff_w2,  ffw216, (size_t)L_ * HID_ * FF_);

    // out = x @ proj_w^T + proj_b
    gemm_wmma_f16<true, false, false, true, false>
        <<<gemm_grid(TB_, HID_), 256, 0, stream>>>(
        a16, proj16, proj_b, out_f, (_Float16*)nullptr, TB_, HID_, IN_);

    for (int i = 0; i < L_; ++i) {
        // h = LN(out) -> f16
        layernorm_f16<<<TB_, 256, 0, stream>>>(
            out_f, ln1_g + (size_t)i * HID_, ln1_b + (size_t)i * HID_, a16, HID_);
        // qkvb = h @ slow_w^T
        gemm_wmma_f16<false, false, false, true, false>
            <<<gemm_grid(TB_, NS_), 256, 0, stream>>>(
            a16, slow16 + (size_t)i * NS_ * HID_, (const float*)nullptr,
            qkvb, (_Float16*)nullptr, TB_, NS_, HID_);
        // softmax(q,k,rk), sigmoid(bt,rbt) in place
        qkvb_post<<<TB_ * H_, 64, 0, stream>>>(qkvb);
        // sequential delta + RNN scans, fused, async-pipelined
        fastrnn_scan<<<B_ * H_, 64, 0, stream>>>(
            qkvb, fw + (size_t)i * B_ * H_ * D_ * D_,
            rfw + (size_t)i * B_ * H_ * D_ * D_,
            rnn + (size_t)i * B_ * H_ * D_, hseq);
        // out += hseq @ out_w^T
        cvt(hseq, a16, (size_t)TB_ * HID_);
        gemm_wmma_f16<false, false, true, true, false>
            <<<gemm_grid(TB_, HID_), 256, 0, stream>>>(
            a16, outw16 + (size_t)i * HID_ * HID_, (const float*)nullptr,
            out_f, (_Float16*)nullptr, TB_, HID_, HID_);
        // FF block: f = LN(out); out += relu(f W1^T + b1) W2^T + b2
        layernorm_f16<<<TB_, 256, 0, stream>>>(
            out_f, ff_ln_g + (size_t)i * HID_, ff_ln_b + (size_t)i * HID_, a16, HID_);
        gemm_wmma_f16<true, true, false, false, true>
            <<<gemm_grid(TB_, FF_), 256, 0, stream>>>(
            a16, ffw116 + (size_t)i * FF_ * HID_, ff_b1 + (size_t)i * FF_,
            (float*)nullptr, ff16, TB_, FF_, HID_);
        gemm_wmma_f16<true, false, true, true, false>
            <<<gemm_grid(TB_, HID_), 256, 0, stream>>>(
            ff16, ffw216 + (size_t)i * HID_ * FF_, ff_b2 + (size_t)i * HID_,
            out_f, (_Float16*)nullptr, TB_, HID_, FF_);
    }

    copy_f32_k<<<dim3((TB_ * HID_ + 255) / 256), 256, 0, stream>>>(
        out_f, (float*)d_out, TB_ * HID_);
}
